// MeshGraphNet_34162169872713
// MI455X (gfx1250) — compile-verified, matrix operand order checked by reading, not run on hardware
//
#include <hip/hip_runtime.h>
#include <hip/hip_bf16.h>

typedef __attribute__((ext_vector_type(16))) _Float16 v16h;
typedef __attribute__((ext_vector_type(8)))  _Float16 v8h;
typedef __attribute__((ext_vector_type(8)))  float    v8f;

// Global (addrspace(1)) views to force global_load/global_store selection.
typedef const __attribute__((address_space(1))) v8h      gc_v8h;
typedef const __attribute__((address_space(1))) float    gc_f32;
typedef __attribute__((address_space(1))) _Float16       g_f16;
typedef __attribute__((address_space(1))) float          g_f32;

#define EPI_LRELU 0
#define EPI_F32   1
#define EPI_DEC   2

// Load a per-lane WMMA 16-bit operand fragment from a precomputed lane pointer.
// Pointer layout per ISA: lane = 16*half + idx holds K [base..base+7] and
// [base+16..base+23] where base = k0 + half*8 within the row.
__device__ __forceinline__ v16h frag_at(const _Float16* __restrict__ p) {
    gc_v8h* gp = (gc_v8h*)p;
    v8h lo = gp[0];
    v8h hi = gp[2];   // +16 halfs = +32 bytes = 2 x v8h
    return __builtin_shufflevector(lo, hi, 0,1,2,3,4,5,6,7,8,9,10,11,12,13,14,15);
}

// C = A[M,K] * W[K,N] + bias, W supplied pre-transposed as WT[N,K] (f16).
// Block = 4 waves (2x2) tiling 128x64 of C; each wave computes a 64x32 tile as
// 4x2 WMMA 16x16x32 subtiles (fully unrolled, constant-indexed accumulators:
// 8 x v8f = 64 VGPRs -> no spills).
// M % 16 == 0, K % 32 == 0, N % 16 == 0 required; M >= 16, N >= 16.
template<int EPI>
__global__ void __launch_bounds__(128, 1)
gemm_wmma(const _Float16* __restrict__ A, const _Float16* __restrict__ WT,
          const float* __restrict__ bias, void* __restrict__ out,
          int M, int K, int N, int outc) {
    const int lane = threadIdx.x & 31;
    const int wave = threadIdx.x >> 5;
    const int idx  = lane & 15;
    const int half = lane >> 4;

    const long m0 = (long)blockIdx.x * 128 + (wave & 1) * 64;
    const int  n0 = blockIdx.y * 64 + (wave >> 1) * 32;
    if (n0 >= N || m0 >= M) return;

    // Clamped subtile bases (tail blocks recompute the last valid subtile).
    long mb[4]; int nb[2];
    #pragma unroll
    for (int i = 0; i < 4; ++i) {
        long mr = m0 + 16 * i;  mb[i] = (mr + 16 <= M) ? mr : (long)(M - 16);
    }
    #pragma unroll
    for (int j = 0; j < 2; ++j) {
        int nr = n0 + 16 * j;   nb[j] = (nr + 16 <= N) ? nr : (N - 16);
    }

    // Hoisted lane pointers, bumped by 32 halfs per K step.
    const _Float16* ap[4]; const _Float16* bp[2];
    #pragma unroll
    for (int i = 0; i < 4; ++i) ap[i] = A  + (mb[i] + idx) * (long)K + half * 8;
    #pragma unroll
    for (int j = 0; j < 2; ++j) bp[j] = WT + (nb[j] + idx) * (long)K + half * 8;

    v8f z = {0.f,0.f,0.f,0.f,0.f,0.f,0.f,0.f};
    v8f acc[4][2];
    #pragma unroll
    for (int i = 0; i < 4; ++i)
        #pragma unroll
        for (int j = 0; j < 2; ++j) acc[i][j] = z;

    for (int k0 = 0; k0 < K; k0 += 32) {
        v16h bf0 = frag_at(bp[0]);
        v16h bf1 = frag_at(bp[1]);
        #pragma unroll
        for (int i = 0; i < 4; ++i) {
            v16h af = frag_at(ap[i]);
            acc[i][0] = __builtin_amdgcn_wmma_f32_16x16x32_f16(
                false, af, false, bf0, (short)0, acc[i][0], false, false);
            acc[i][1] = __builtin_amdgcn_wmma_f32_16x16x32_f16(
                false, af, false, bf1, (short)0, acc[i][1], false, false);
            ap[i] += 32;
        }
        bp[0] += 32; bp[1] += 32;
    }

    gc_f32* gbias  = (gc_f32*)bias;
    g_f16*  out16  = (g_f16*)out;
    g_f32*  out32  = (g_f32*)out;

    #pragma unroll
    for (int i = 0; i < 4; ++i) {
        if (m0 + 16 * i < M) {
            const long mrow = m0 + 16 * i + 8 * half;  // D: VGPR r -> M = r + 8*half
            #pragma unroll
            for (int j = 0; j < 2; ++j) {
                if (n0 + 16 * j < N) {
                    const int col = n0 + 16 * j + idx;
                    float bv;
                    if (EPI == EPI_DEC) bv = (col < outc) ? gbias[col] : 0.f;
                    else                bv = gbias[col];
                    #pragma unroll
                    for (int r = 0; r < 8; ++r) {
                        float y = acc[i][j][r] + bv;
                        long m = mrow + r;
                        if (EPI == EPI_LRELU) {
                            y = (y > 0.f) ? y : 0.01f * y;
                            out16[m * (long)N + col] = (_Float16)y;
                        } else if (EPI == EPI_F32) {
                            out32[m * (long)N + col] = y;
                        } else {
                            if (col < outc) out32[m * (long)outc + col] = y;
                        }
                    }
                }
            }
        }
    }
}

// Convert fp32 W[Ksrc,Nsrc] (row-major) -> f16 WT[Npad,Kpad] (row-major), zero-padded.
__global__ void wt_k(const float* __restrict__ W, _Float16* __restrict__ WT,
                     int Ksrc, int Kpad, int Nsrc, int Npad) {
    long i = blockIdx.x * (long)blockDim.x + threadIdx.x;
    long tot = (long)Npad * Kpad;
    if (i >= tot) return;
    int n = (int)(i / Kpad), k = (int)(i % Kpad);
    float v = (n < Nsrc && k < Ksrc) ? W[(long)k * Nsrc + n] : 0.f;
    WT[i] = (_Float16)v;
}

// fp32 [rows,fin] -> f16 [rows,kpad], zero-padded columns. block = kpad threads.
__global__ void pack_k(const float* __restrict__ in, _Float16* __restrict__ out,
                       int fin, int kpad) {
    long i = blockIdx.x;
    int t = threadIdx.x;
    out[i * (long)kpad + t] = (t < fin) ? (_Float16)in[i * (long)fin + t] : (_Float16)0.f;
}

// LayerNorm over width 128 (+ optional residual), dual f32/f16 output. block=128.
__global__ void layernorm_k(const float* __restrict__ pre, const float* __restrict__ g,
                            const float* __restrict__ be, const float* __restrict__ res,
                            float* __restrict__ out32, _Float16* __restrict__ out16) {
    __shared__ float red[128];
    const long row = blockIdx.x;
    const int t = threadIdx.x;
    float x = pre[row * 128 + t];
    red[t] = x; __syncthreads();
    for (int s = 64; s > 0; s >>= 1) { if (t < s) red[t] += red[t + s]; __syncthreads(); }
    float mu = red[0] * (1.f / 128.f);
    __syncthreads();
    float d = x - mu;
    red[t] = d * d; __syncthreads();
    for (int s = 64; s > 0; s >>= 1) { if (t < s) red[t] += red[t + s]; __syncthreads(); }
    float var = red[0] * (1.f / 128.f);
    float y = d * rsqrtf(var + 1e-5f) * g[t] + be[t];
    if (res) y += res[row * 128 + t];
    out32[row * 128 + t] = y;
    out16[row * 128 + t] = (_Float16)y;
}

// cat_e[e] = [edge16[e] | node16[src[e]] | node16[dst[e]]], widths 128 each. block=128.
__global__ void concat_e_k(const _Float16* __restrict__ edge16, const _Float16* __restrict__ node16,
                           const int* __restrict__ src, const int* __restrict__ dst,
                           _Float16* __restrict__ cat) {
    long e = blockIdx.x;
    int t = threadIdx.x;
    long s = src[e], d = dst[e];
    cat[e * 384 + t]       = edge16[e * 128 + t];
    cat[e * 384 + 128 + t] = node16[s * 128 + t];
    cat[e * 384 + 256 + t] = node16[d * 128 + t];
}

// cat_n[i] = [node16[i] | f16(pe_sum[i])]. block=256.
__global__ void concat_n_k(const _Float16* __restrict__ node16, const float* __restrict__ pe,
                           _Float16* __restrict__ cat) {
    long i = blockIdx.x;
    int t = threadIdx.x;
    cat[i * 256 + t] = (t < 128) ? node16[i * 128 + t]
                                 : (_Float16)pe[i * 128 + (t - 128)];
}

__global__ void zero_k(float* __restrict__ p, long n) {
    long i = blockIdx.x * (long)blockDim.x + threadIdx.x;
    if (i < n) p[i] = 0.f;
}

// segment_sum: pe_sum[dst[e]] += edge32[e] (fp32 global atomics).
__global__ void scatter_add_k(const float* __restrict__ edge32, const int* __restrict__ dst,
                              float* __restrict__ pe, long total) {
    long i = blockIdx.x * (long)blockDim.x + threadIdx.x;
    if (i >= total) return;
    long e = i >> 7;
    int c = (int)(i & 127);
    atomicAdd(&pe[(long)dst[e] * 128 + c], edge32[i]);
}

extern "C" void kernel_launch(void* const* d_in, const int* in_sizes, int n_in,
                              void* d_out, int out_size, void* d_ws, size_t ws_size,
                              hipStream_t stream) {
    (void)n_in; (void)out_size; (void)ws_size;
    const int N = in_sizes[0] / 16;   // 20000
    const int E = in_sizes[1] / 5;    // 100000
    const int P = 4, H = 2;

    int a = 0;
    const float* nfeat = (const float*)d_in[a++];
    const float* efeat = (const float*)d_in[a++];
    const int*   src   = (const int*)d_in[a++];
    const int*   dst   = (const int*)d_in[a++];
    #define NXT() ((const float*)d_in[a++])
    const float *en_Wi=NXT(), *en_bi=NXT(), *en_Wh=NXT(), *en_bh=NXT(), *en_Wo=NXT(), *en_bo=NXT(), *en_g=NXT(), *en_be=NXT();
    const float *ee_Wi=NXT(), *ee_bi=NXT(), *ee_Wh=NXT(), *ee_bh=NXT(), *ee_Wo=NXT(), *ee_bo=NXT(), *ee_g=NXT(), *ee_be=NXT();
    const float *pe_Wi=NXT(), *pe_bi=NXT(), *pe_Wh=NXT(), *pe_bh=NXT(), *pe_Wo=NXT(), *pe_bo=NXT(), *pe_g=NXT(), *pe_be=NXT();
    const float *pn_Wi=NXT(), *pn_bi=NXT(), *pn_Wh=NXT(), *pn_bh=NXT(), *pn_Wo=NXT(), *pn_bo=NXT(), *pn_g=NXT(), *pn_be=NXT();
    const float *dc_Wi=NXT(), *dc_bi=NXT(), *dc_Wh=NXT(), *dc_bh=NXT(), *dc_Wo=NXT(), *dc_bo=NXT();
    #undef NXT

    // ---- workspace bump allocator (256B aligned) ----
    char* ws = (char*)d_ws;
    size_t off = 0;
    auto alloc = [&](size_t bytes) -> void* {
        off = (off + 255) & ~(size_t)255;
        void* p = ws + off;
        off += bytes;
        return p;
    };
    auto allocH = [&](size_t elems) -> _Float16* { return (_Float16*)alloc(elems * 2); };
    auto allocF = [&](size_t elems) -> float*    { return (float*)alloc(elems * 4); };

    auto conv = [&](const float* W, _Float16* WT, int Ks, int Kp, int Ns, int Np) {
        long tot = (long)Np * Kp;
        wt_k<<<(unsigned)((tot + 255) / 256), 256, 0, stream>>>(W, WT, Ks, Kp, Ns, Np);
    };

    // ---- f16 transposed weights ----
    _Float16 *enWiT = allocH(256 * 32),  *enWhT[2], *enWoT = allocH(128 * 256);
    _Float16 *eeWiT = allocH(256 * 32),  *eeWhT[2], *eeWoT = allocH(128 * 256);
    _Float16 *peWiT[4], *peWhT[4][2], *peWoT[4];
    _Float16 *pnWiT[4], *pnWhT[4][2], *pnWoT[4];
    _Float16 *dcWiT = allocH(256 * 128), *dcWhT[2], *dcWoT = allocH(16 * 256);
    for (int j = 0; j < H; ++j) { enWhT[j] = allocH(65536); eeWhT[j] = allocH(65536); dcWhT[j] = allocH(65536); }
    for (int i = 0; i < P; ++i) {
        peWiT[i] = allocH(256 * 384); peWoT[i] = allocH(128 * 256);
        pnWiT[i] = allocH(256 * 256); pnWoT[i] = allocH(128 * 256);
        for (int j = 0; j < H; ++j) { peWhT[i][j] = allocH(65536); pnWhT[i][j] = allocH(65536); }
    }

    conv(en_Wi, enWiT, 16, 32, 256, 256);
    conv(ee_Wi, eeWiT,  5, 32, 256, 256);
    conv(en_Wo, enWoT, 256, 256, 128, 128);
    conv(ee_Wo, eeWoT, 256, 256, 128, 128);
    conv(dc_Wi, dcWiT, 128, 128, 256, 256);
    conv(dc_Wo, dcWoT, 256, 256, 2, 16);
    for (int j = 0; j < H; ++j) {
        conv(en_Wh + (long)j * 65536, enWhT[j], 256, 256, 256, 256);
        conv(ee_Wh + (long)j * 65536, eeWhT[j], 256, 256, 256, 256);
        conv(dc_Wh + (long)j * 65536, dcWhT[j], 256, 256, 256, 256);
    }
    for (int i = 0; i < P; ++i) {
        conv(pe_Wi + (long)i * 384 * 256, peWiT[i], 384, 384, 256, 256);
        conv(pn_Wi + (long)i * 256 * 256, pnWiT[i], 256, 256, 256, 256);
        conv(pe_Wo + (long)i * 256 * 128, peWoT[i], 256, 256, 128, 128);
        conv(pn_Wo + (long)i * 256 * 128, pnWoT[i], 256, 256, 128, 128);
        for (int j = 0; j < H; ++j) {
            conv(pe_Wh + ((long)i * 2 + j) * 65536, peWhT[i][j], 256, 256, 256, 256);
            conv(pn_Wh + ((long)i * 2 + j) * 65536, pnWhT[i][j], 256, 256, 256, 256);
        }
    }

    // ---- activations / state ----
    _Float16* in_n   = allocH((size_t)N * 32);
    _Float16* in_e   = allocH((size_t)E * 32);
    _Float16* act0   = allocH((size_t)E * 256);
    _Float16* act1   = allocH((size_t)E * 256);
    _Float16* cat_e  = allocH((size_t)E * 384);
    _Float16* cat_n  = allocH((size_t)N * 256);
    float*    pre    = allocF((size_t)E * 128);
    float*    node32 = allocF((size_t)N * 128);
    _Float16* node16 = allocH((size_t)N * 128);
    float*    edge32 = allocF((size_t)E * 128);
    _Float16* edge16 = allocH((size_t)E * 128);
    float*    pe_sum = allocF((size_t)N * 128);

    auto gemm = [&](const _Float16* A, const _Float16* WT, const float* bias, void* out,
                    int M, int K, int Nn, int epi, int outc) {
        dim3 grid((M + 127) / 128, (Nn + 63) / 64);
        dim3 blk(128);
        if (epi == EPI_LRELU)
            gemm_wmma<EPI_LRELU><<<grid, blk, 0, stream>>>(A, WT, bias, out, M, K, Nn, outc);
        else if (epi == EPI_F32)
            gemm_wmma<EPI_F32><<<grid, blk, 0, stream>>>(A, WT, bias, out, M, K, Nn, outc);
        else
            gemm_wmma<EPI_DEC><<<grid, blk, 0, stream>>>(A, WT, bias, out, M, K, Nn, outc);
    };

    auto mlp = [&](const _Float16* X, int M, int Kin,
                   const _Float16* WiT, const float* bi,
                   _Float16* const* WhT, const float* bh,
                   const _Float16* WoT, const float* bo) {
        gemm(X,    WiT,    bi,       act0, M, Kin, 256, EPI_LRELU, 0);
        gemm(act0, WhT[0], bh,       act1, M, 256, 256, EPI_LRELU, 0);
        gemm(act1, WhT[1], bh + 256, act0, M, 256, 256, EPI_LRELU, 0);
        gemm(act0, WoT,    bo,       pre,  M, 256, 128, EPI_F32,   0);
    };

    // ---- encoders ----
    pack_k<<<N, 32, 0, stream>>>(nfeat, in_n, 16, 32);
    pack_k<<<E, 32, 0, stream>>>(efeat, in_e, 5, 32);

    mlp(in_n, N, 32, enWiT, en_bi, enWhT, en_bh, enWoT, en_bo);
    layernorm_k<<<N, 128, 0, stream>>>(pre, en_g, en_be, nullptr, node32, node16);

    mlp(in_e, E, 32, eeWiT, ee_bi, eeWhT, ee_bh, eeWoT, ee_bo);
    layernorm_k<<<E, 128, 0, stream>>>(pre, ee_g, ee_be, nullptr, edge32, edge16);

    // ---- processor passes ----
    for (int i = 0; i < P; ++i) {
        concat_e_k<<<E, 128, 0, stream>>>(edge16, node16, src, dst, cat_e);
        mlp(cat_e, E, 384, peWiT[i], pe_bi + (long)i * 256, peWhT[i], pe_bh + (long)i * 512,
            peWoT[i], pe_bo + (long)i * 128);
        layernorm_k<<<E, 128, 0, stream>>>(pre, pe_g + (long)i * 128, pe_be + (long)i * 128,
                                           edge32, edge32, edge16);

        long nn = (long)N * 128;
        zero_k<<<(unsigned)((nn + 255) / 256), 256, 0, stream>>>(pe_sum, nn);
        long tot = (long)E * 128;
        scatter_add_k<<<(unsigned)((tot + 255) / 256), 256, 0, stream>>>(edge32, dst, pe_sum, tot);

        concat_n_k<<<N, 256, 0, stream>>>(node16, pe_sum, cat_n);
        mlp(cat_n, N, 256, pnWiT[i], pn_bi + (long)i * 256, pnWhT[i], pn_bh + (long)i * 512,
            pnWoT[i], pn_bo + (long)i * 128);
        layernorm_k<<<N, 128, 0, stream>>>(pre, pn_g + (long)i * 128, pn_be + (long)i * 128,
                                           node32, node32, node16);
    }

    // ---- decoder ----
    gemm(node16, dcWiT,    dc_bi,       act0,  N, 128, 256, EPI_LRELU, 0);
    gemm(act0,   dcWhT[0], dc_bh,       act1,  N, 256, 256, EPI_LRELU, 0);
    gemm(act1,   dcWhT[1], dc_bh + 256, act0,  N, 256, 256, EPI_LRELU, 0);
    gemm(act0,   dcWoT,    dc_bo,       d_out, N, 256, 16,  EPI_DEC,   2);
}
